// SelfAttention2d_15229954032019
// MI455X (gfx1250) — compile-verified, matrix operand order checked by reading, not run on hardware
//
#include <hip/hip_runtime.h>
#include <hip/hip_bf16.h>
#include <stdint.h>

// ---------------------------------------------------------------------------
// SelfAttention2d for MI455X (gfx1250, wave32, WMMA).
// B=8, C=256, N=H*W=4096, Ck=32.
//  k_proj : f,g,h = w{f,g,h} @ x ; store fT,gT as [N,32] bf16, h as [32,N] bf16
//  k_stats: rowmax/rowinv of softmax(s, axis=m); 4 n-tiles per wave so each
//           gT B-tile load feeds 4 v_wmma (register blocking -> 4x less L2)
//  k_attnv: vT[m,c] = sum_n beta^T[m,n] h^T[n,c]; 2 m-tiles per wave; s
//           recomputed by WMMA, beta repacked lane-locally (C/D layout of s
//           == A layout of beta^T) -> 8 WMMAs per 32-n step
//  k_out  : out = gamma * (wv @ v) + x
// ---------------------------------------------------------------------------

typedef __attribute__((ext_vector_type(16))) __bf16 v16bf;
typedef __attribute__((ext_vector_type(8)))  float  v8f;

#define B_  8
#define C_  256
#define N_  4096
#define CK_ 32

__device__ __forceinline__ unsigned short f2bf(float f) {
  union { float f; unsigned u; } v; v.f = f;
  unsigned r = v.u + 0x7FFFu + ((v.u >> 16) & 1u);   // round-to-nearest-even
  return (unsigned short)(r >> 16);
}

// Load a 16-half A/B operand per ISA 16-bit layout: this lane needs elements
// [klo, klo+8) and [klo+16, klo+24) of a 32-element K row (p points at +klo).
__device__ __forceinline__ v16bf ld_tile(const unsigned short* p) {
  union { v16bf v; uint4 u[2]; } t;
  t.u[0] = *(const uint4*)(p);        // K = klo .. klo+7
  t.u[1] = *(const uint4*)(p + 16);   // K = klo+16 .. klo+23
  return t.v;
}

__device__ __forceinline__ v8f wmma_bf16(v16bf a, v16bf b, v8f c) {
  return __builtin_amdgcn_wmma_f32_16x16x32_bf16(
      /*neg_a=*/false, a, /*neg_b=*/false, b,
      /*c_mod=*/(short)0, c, /*reuse_a=*/false, /*reuse_b=*/false);
}

__device__ __forceinline__ void ld8f(const float* p, float* d) {
  float4 a = *(const float4*)(p);
  float4 b = *(const float4*)(p + 4);
  d[0] = a.x; d[1] = a.y; d[2] = a.z; d[3] = a.w;
  d[4] = b.x; d[5] = b.y; d[6] = b.z; d[7] = b.w;
}

// ---------------------------------------------------------------------------
// Kernel 1: projections f,g,h (K=256 reduction, uniform scalar weights).
// ---------------------------------------------------------------------------
__global__ void __launch_bounds__(256) k_proj(
    const float* __restrict__ x,  const float* __restrict__ wf,
    const float* __restrict__ wg, const float* __restrict__ wh,
    unsigned short* __restrict__ fT, unsigned short* __restrict__ gT,
    unsigned short* __restrict__ hM)
{
  int tid = blockIdx.x * blockDim.x + threadIdx.x;   // over B*N
  int b = tid >> 12;
  int n = tid & (N_ - 1);
  const float* xb = x + (size_t)b * C_ * N_ + n;

  float fa[CK_], ga[CK_], ha[CK_];
#pragma unroll
  for (int o = 0; o < CK_; ++o) { fa[o] = 0.f; ga[o] = 0.f; ha[o] = 0.f; }

  for (int c = 0; c < C_; ++c) {
    float xv = xb[(size_t)c * N_];
#pragma unroll
    for (int o = 0; o < CK_; ++o) {
      fa[o] = __builtin_fmaf(wf[o * C_ + c], xv, fa[o]);
      ga[o] = __builtin_fmaf(wg[o * C_ + c], xv, ga[o]);
      ha[o] = __builtin_fmaf(wh[o * C_ + c], xv, ha[o]);
    }
  }

  unsigned short* fr = fT + ((size_t)b * N_ + n) * CK_;   // fT: [N,32] row n
  unsigned short* gr = gT + ((size_t)b * N_ + n) * CK_;   // gT: [N,32] row n
  unsigned short* hb = hM + (size_t)b * CK_ * N_ + n;     // h : [32,N]
#pragma unroll
  for (int o = 0; o < CK_; ++o) {
    fr[o] = f2bf(fa[o]);
    gr[o] = f2bf(ga[o]);
    hb[(size_t)o * N_] = f2bf(ha[o]);
  }
}

// ---------------------------------------------------------------------------
// Kernel 2: softmax row stats over m. One wave owns FOUR 16-row n-tiles
// (A tiles stay in registers); each streamed gT B-tile feeds 4 WMMAs.
// Sweep 1: per-lane running max, xor-reduce (masks 1..8 stay in each half).
// Sweep 2: sum of exp(s - max), xor-reduce. Store rowmax + 1/rowsum.
// ---------------------------------------------------------------------------
#define NT_ 4
__global__ void __launch_bounds__(128) k_stats(
    const unsigned short* __restrict__ fT, const unsigned short* __restrict__ gT,
    float* __restrict__ rowmax, float* __restrict__ rowinv)
{
  int wid  = (blockIdx.x * blockDim.x + threadIdx.x) >> 5;  // 0..B*(N/64)-1
  int lane = threadIdx.x & 31;
  int b    = wid >> 6;            // 64 row-groups (of 64 rows) per batch
  int ng   = wid & 63;
  int r    = lane & 15;
  int klo  = (lane >> 4) * 8;

  const unsigned short* fb = fT + (size_t)b * N_ * CK_;
  const unsigned short* gb = gT + (size_t)b * N_ * CK_;

  v16bf A[NT_];
#pragma unroll
  for (int t = 0; t < NT_; ++t)
    A[t] = ld_tile(fb + (size_t)(ng * 64 + t * 16 + r) * CK_ + klo);

  float mx[NT_][8];
#pragma unroll
  for (int t = 0; t < NT_; ++t)
#pragma unroll
    for (int i = 0; i < 8; ++i) mx[t][i] = -3.4e38f;

  for (int mt = 0; mt < N_ / 16; ++mt) {
    v16bf Bm = ld_tile(gb + (size_t)(mt * 16 + r) * CK_ + klo);
#pragma unroll
    for (int t = 0; t < NT_; ++t) {
      v8f cz = {0.f, 0.f, 0.f, 0.f, 0.f, 0.f, 0.f, 0.f};
      v8f s = wmma_bf16(A[t], Bm, cz);
#pragma unroll
      for (int i = 0; i < 8; ++i) mx[t][i] = fmaxf(mx[t][i], s[i]);
    }
  }
#pragma unroll
  for (int m = 1; m <= 8; m <<= 1)
#pragma unroll
    for (int t = 0; t < NT_; ++t)
#pragma unroll
      for (int i = 0; i < 8; ++i)
        mx[t][i] = fmaxf(mx[t][i], __shfl_xor(mx[t][i], m, 32));

  float sm[NT_][8];
#pragma unroll
  for (int t = 0; t < NT_; ++t)
#pragma unroll
    for (int i = 0; i < 8; ++i) sm[t][i] = 0.f;

  for (int mt = 0; mt < N_ / 16; ++mt) {
    v16bf Bm = ld_tile(gb + (size_t)(mt * 16 + r) * CK_ + klo);
#pragma unroll
    for (int t = 0; t < NT_; ++t) {
      v8f cz = {0.f, 0.f, 0.f, 0.f, 0.f, 0.f, 0.f, 0.f};
      v8f s = wmma_bf16(A[t], Bm, cz);
#pragma unroll
      for (int i = 0; i < 8; ++i) sm[t][i] += __expf(s[i] - mx[t][i]);
    }
  }
#pragma unroll
  for (int m = 1; m <= 8; m <<= 1)
#pragma unroll
    for (int t = 0; t < NT_; ++t)
#pragma unroll
      for (int i = 0; i < 8; ++i) sm[t][i] += __shfl_xor(sm[t][i], m, 32);

  if (r == 0) {  // lanes 0 and 16 (klo = 0 / 8) cover n-rows 0..7 / 8..15
#pragma unroll
    for (int t = 0; t < NT_; ++t) {
      int nb = b * N_ + ng * 64 + t * 16 + klo;
#pragma unroll
      for (int i = 0; i < 8; ++i) {
        rowmax[nb + i] = mx[t][i];
        rowinv[nb + i] = 1.0f / sm[t][i];
      }
    }
  }
}

// ---------------------------------------------------------------------------
// Kernel 3: vT[m,c] = sum_n beta^T[m,n] * h^T[n,c]. One wave owns TWO m-tiles
// (Bm0/Bm1 resident), so fT / h / stats loads are amortized over 8 WMMAs per
// 32-n step: 4 s-tile WMMAs + 4 accumulate WMMAs.
// ---------------------------------------------------------------------------
__global__ void __launch_bounds__(128) k_attnv(
    const unsigned short* __restrict__ fT, const unsigned short* __restrict__ gT,
    const unsigned short* __restrict__ hM, const float* __restrict__ rowmax,
    const float* __restrict__ rowinv, float* __restrict__ vT)
{
  int wid  = (blockIdx.x * blockDim.x + threadIdx.x) >> 5;  // 0..B*(N/32)-1
  int lane = threadIdx.x & 31;
  int b    = wid >> 7;            // 128 col-groups (of 32 cols) per batch
  int mg   = wid & 127;
  int r    = lane & 15;
  int klo  = (lane >> 4) * 8;

  const unsigned short* fb = fT + (size_t)b * N_ * CK_;
  const unsigned short* gb = gT + (size_t)b * N_ * CK_;
  const unsigned short* hb = hM + (size_t)b * CK_ * N_;
  const float* mxp = rowmax + (size_t)b * N_;
  const float* ivp = rowinv + (size_t)b * N_;

  // B operands of the s-GEMM: gT columns m = mg*32 .. mg*32+31 (two tiles)
  v16bf Bm0 = ld_tile(gb + (size_t)(mg * 32 + r) * CK_ + klo);
  v16bf Bm1 = ld_tile(gb + (size_t)(mg * 32 + 16 + r) * CK_ + klo);

  v8f acc00 = {0.f,0.f,0.f,0.f,0.f,0.f,0.f,0.f};  // m-tile 0, c = 0..15
  v8f acc01 = {0.f,0.f,0.f,0.f,0.f,0.f,0.f,0.f};  // m-tile 0, c = 16..31
  v8f acc10 = {0.f,0.f,0.f,0.f,0.f,0.f,0.f,0.f};  // m-tile 1, c = 0..15
  v8f acc11 = {0.f,0.f,0.f,0.f,0.f,0.f,0.f,0.f};  // m-tile 1, c = 16..31

  for (int nb = 0; nb < N_; nb += 32) {
    v16bf A0 = ld_tile(fb + (size_t)(nb + r) * CK_ + klo);        // n = nb..nb+15
    v16bf A1 = ld_tile(fb + (size_t)(nb + 16 + r) * CK_ + klo);   // n = nb+16..nb+31

    // B operands of the v-GEMM: h^T[n = nb..nb+31, c]; h stored [32,N]
    v16bf H0 = ld_tile(hb + (size_t)r * N_ + nb + klo);           // c = 0..15
    v16bf H1 = ld_tile(hb + (size_t)(16 + r) * N_ + nb + klo);    // c = 16..31

    // Softmax stats for this lane's n indices (shared by both m-tiles):
    // tile0 -> nb+klo+i, tile1 -> nb+klo+16+i
    int no = nb + klo;
    float mx0[8], mx1[8], iv0[8], iv1[8];
    ld8f(mxp + no,      mx0);  ld8f(mxp + no + 16, mx1);
    ld8f(ivp + no,      iv0);  ld8f(ivp + no + 16, iv1);

    v8f cz = {0.f,0.f,0.f,0.f,0.f,0.f,0.f,0.f};

    // ---- m-tile 0 ----
    {
      v8f s0 = wmma_bf16(A0, Bm0, cz);
      v8f s1 = wmma_bf16(A1, Bm0, cz);
      // C/D layout of s == A layout of beta^T: halfs[0..7]=tile0, [8..15]=tile1
      union { v16bf v; unsigned short u[16]; } pk;
#pragma unroll
      for (int i = 0; i < 8; ++i) {
        pk.u[i]     = f2bf(__expf(s0[i] - mx0[i]) * iv0[i]);
        pk.u[8 + i] = f2bf(__expf(s1[i] - mx1[i]) * iv1[i]);
      }
      acc00 = wmma_bf16(pk.v, H0, acc00);
      acc01 = wmma_bf16(pk.v, H1, acc01);
    }
    // ---- m-tile 1 ----
    {
      v8f s0 = wmma_bf16(A0, Bm1, cz);
      v8f s1 = wmma_bf16(A1, Bm1, cz);
      union { v16bf v; unsigned short u[16]; } pk;
#pragma unroll
      for (int i = 0; i < 8; ++i) {
        pk.u[i]     = f2bf(__expf(s0[i] - mx0[i]) * iv0[i]);
        pk.u[8 + i] = f2bf(__expf(s1[i] - mx1[i]) * iv1[i]);
      }
      acc10 = wmma_bf16(pk.v, H0, acc10);
      acc11 = wmma_bf16(pk.v, H1, acc11);
    }
  }

  // D layout: lane r (+half) holds vT[m = mbase + klo + i, c = r (+16)]
  {
    int mrow = mg * 32 + klo;
    float* vrow = vT + ((size_t)b * N_ + mrow) * CK_ + r;
#pragma unroll
    for (int i = 0; i < 8; ++i) {
      vrow[(size_t)i * CK_]      = acc00[i];
      vrow[(size_t)i * CK_ + 16] = acc01[i];
    }
  }
  {
    int mrow = mg * 32 + 16 + klo;
    float* vrow = vT + ((size_t)b * N_ + mrow) * CK_ + r;
#pragma unroll
    for (int i = 0; i < 8; ++i) {
      vrow[(size_t)i * CK_]      = acc10[i];
      vrow[(size_t)i * CK_ + 16] = acc11[i];
    }
  }
}

// ---------------------------------------------------------------------------
// Kernel 4: out = gamma * (wv @ v) + x. Coalesced over n; wv row is uniform
// per block (block spans one (b,c) row) -> scalar loads.
// ---------------------------------------------------------------------------
__global__ void __launch_bounds__(256) k_out(
    const float* __restrict__ vT, const float* __restrict__ wv,
    const float* __restrict__ x,  const float* __restrict__ gamma,
    float* __restrict__ out)
{
  size_t tid = (size_t)blockIdx.x * blockDim.x + threadIdx.x;  // over B*C*N
  int n = (int)(tid & (N_ - 1));
  int c = (int)((tid >> 12) & (C_ - 1));
  int b = (int)(tid >> 20);

  const float* vr = vT + ((size_t)b * N_ + n) * CK_;
  const float* wr = wv + (size_t)c * CK_;
  float acc = 0.f;
#pragma unroll
  for (int k = 0; k < CK_; ++k) acc = __builtin_fmaf(wr[k], vr[k], acc);
  out[tid] = gamma[0] * acc + x[tid];
}

// ---------------------------------------------------------------------------
extern "C" void kernel_launch(void* const* d_in, const int* in_sizes, int n_in,
                              void* d_out, int out_size, void* d_ws, size_t ws_size,
                              hipStream_t stream)
{
  (void)in_sizes; (void)n_in; (void)out_size; (void)ws_size;
  const float* x     = (const float*)d_in[0];
  const float* wf    = (const float*)d_in[1];
  const float* wg    = (const float*)d_in[2];
  const float* wh    = (const float*)d_in[3];
  const float* wv    = (const float*)d_in[4];
  const float* gamma = (const float*)d_in[5];
  float* out = (float*)d_out;

  char* ws = (char*)d_ws;
  size_t off = 0;
  auto carve = [&](size_t bytes) -> void* {
    void* p = ws + off;
    off += (bytes + 255) & ~(size_t)255;
    return p;
  };
  unsigned short* fT = (unsigned short*)carve((size_t)B_ * N_ * CK_ * 2);  // [B][N][32] bf16
  unsigned short* gT = (unsigned short*)carve((size_t)B_ * N_ * CK_ * 2);  // [B][N][32] bf16
  unsigned short* hM = (unsigned short*)carve((size_t)B_ * CK_ * N_ * 2);  // [B][32][N] bf16
  float* rowmax = (float*)carve((size_t)B_ * N_ * 4);
  float* rowinv = (float*)carve((size_t)B_ * N_ * 4);
  float* vT     = (float*)carve((size_t)B_ * N_ * CK_ * 4);               // [B][N][32] f32

  k_proj <<<(B_ * N_) / 256, 256, 0, stream>>>(x, wf, wg, wh, fT, gT, hM);
  k_stats<<<(B_ * (N_ / 64)) / 4, 128, 0, stream>>>(fT, gT, rowmax, rowinv);
  k_attnv<<<(B_ * (N_ / 32)) / 4, 128, 0, stream>>>(fT, gT, hM, rowmax, rowinv, vT);
  k_out  <<<(B_ * C_ * N_) / 256, 256, 0, stream>>>(vT, wv, x, gamma, out);
}
